// GraphConvSparseOne_77206332113740
// MI455X (gfx1250) — matrix operand order repaired and checked by reading, not compiled, not run on hardware
//
#include <hip/hip_runtime.h>

typedef __attribute__((ext_vector_type(16))) _Float16 v16h;
typedef __attribute__((ext_vector_type(8)))  _Float16 v8h;
typedef __attribute__((ext_vector_type(4)))  _Float16 v4h;
typedef __attribute__((ext_vector_type(8)))  float    v8f;
typedef __attribute__((ext_vector_type(4)))  float    v4f;
typedef __attribute__((ext_vector_type(4)))  int      v4i;

#define NN    8192   // nodes
#define FD    128    // features in/out
#define KT    32     // K per WMMA step (f16 16x16x32)
#define TM    128    // rows per block tile
#define APADF 36     // f32 per row of A tile in LDS (32 + 4 pad -> conflict-free b128 reads)
#define YPAD  136    // halves per row of Y tile in LDS (128 + 8 pad)
#define NITER (NN / KT)

// async global->LDS DMA (ASYNCcnt) if the toolchain exposes it
#if defined(__has_builtin)
#if __has_builtin(__builtin_amdgcn_global_load_async_to_lds_b128) && \
    __has_builtin(__builtin_amdgcn_s_wait_asynccnt)
#define USE_ASYNC_LDS 1
#endif
#endif

#ifdef USE_ASYNC_LDS
// builtin expects: (v4i addrspace(1)*, v4i addrspace(3)*, imm offset, imm cpol)
typedef __attribute__((address_space(1))) v4i ga_v4i;
typedef __attribute__((address_space(3))) v4i ls_v4i;
#endif

// ---------------------------------------------------------------------------
// Pack X (f32 [8192,128]) into f16 B-operand fragments, fragment-major:
// frag(kk,nt): 32 lanes x 16 halves; lane l holds column nt*16+(l&15),
// K = kk*32 + (l>>4)*8 + h + (h&8)  (ISA 16-bit B 32x16 layout).
// ---------------------------------------------------------------------------
__global__ __launch_bounds__(256) void pack_x(const float* __restrict__ X,
                                              _Float16* __restrict__ xb) {
  int t   = blockIdx.x * 256 + threadIdx.x;   // 0..65535
  int kk  = t >> 8;                            // 0..255
  int rem = t & 255;
  int nt  = rem >> 5;                          // 0..7
  int l   = rem & 31;
  int col   = nt * 16 + (l & 15);
  int kbase = kk * KT + (l >> 4) * 8;
  v16h v;
#pragma unroll
  for (int h = 0; h < 16; ++h)
    v[h] = (_Float16)X[(size_t)(kbase + h + (h & 8)) * FD + col];
  *(v16h*)(xb + (size_t)t * 16) = v;
}

// Pack a [C,128,128] (or [128,128] with C=1) weight into f16 B fragments.
// t = c*1024 + kk2*256 + nt*32 + l
__global__ __launch_bounds__(256) void pack_w(const float* __restrict__ Wsrc,
                                              _Float16* __restrict__ wb) {
  int t   = blockIdx.x * 256 + threadIdx.x;
  int l   = t & 31;
  int nt  = (t >> 5) & 7;
  int kk2 = (t >> 8) & 3;
  int c   = t >> 10;
  int col   = nt * 16 + (l & 15);
  int kbase = kk2 * KT + (l >> 4) * 8;
  const float* src = Wsrc + (size_t)c * FD * FD;
  v16h v;
#pragma unroll
  for (int h = 0; h < 16; ++h)
    v[h] = (_Float16)src[(size_t)(kbase + h + (h & 8)) * FD + col];
  *(v16h*)(wb + (size_t)t * 16) = v;
}

// ---------------------------------------------------------------------------
// Stage one 128x32 f32 tile of A into LDS buffer (kk % 3):
// async DMA (ASYNCcnt) when available, plain copy otherwise.
// ---------------------------------------------------------------------------
__device__ __forceinline__ void stage_tile(const float* __restrict__ A,
                                           float* __restrict__ ldsA,
                                           int row0, int r0, int jj, int kk) {
  const int k   = kk * KT;
  const int buf = (kk % 3) * (TM * APADF);
#pragma unroll
  for (int i = 0; i < 4; ++i) {
    int r = r0 + i * 32;
    const float* gp = A + (size_t)(row0 + r) * NN + k + jj * 4;
    float* lp = ldsA + buf + r * APADF + jj * 4;
#ifdef USE_ASYNC_LDS
    __builtin_amdgcn_global_load_async_to_lds_b128((ga_v4i*)gp, (ls_v4i*)lp, 0, 0);
#else
    *(v4f*)lp = *(const v4f*)gp;
#endif
    __builtin_prefetch(gp + 4 * KT, 0, 1);   // warm L2 ~4 tiles ahead of DMA
  }
}

// ---------------------------------------------------------------------------
// Main kernel: per (row-tile, channel):
//   Y = A_c[tile,:] @ X      (K = 8192, f16 WMMA/f32 acc, A streamed via
//                             triple-buffered async-DMA LDS pipeline)
//   Z = relu(Y @ adjW_c)     (K = 128, Y transposed through LDS)
//   atomicAdd into accbuf
// ---------------------------------------------------------------------------
__global__ __launch_bounds__(256) void adj_gemm(const float* __restrict__ adjs,
                                                const _Float16* __restrict__ xb,
                                                const _Float16* __restrict__ wab,
                                                float* __restrict__ accbuf) {
  // A triple-buffer: 3*128*36*4 = 55296 B ; Y tile: 128*136*2 = 34816 B (reused)
  __shared__ __align__(16) char smem[3 * TM * APADF * 4];
  float*    ldsA = (float*)smem;
  _Float16* ldsY = (_Float16*)smem;

  const int tid  = threadIdx.x;
  const int lane = tid & 31;
  const int w    = tid >> 5;            // wave 0..7 -> rows [16w,16w+16) of tile
  const int row0 = blockIdx.x * TM;     // 64 row tiles
  const int c    = blockIdx.y;          // 4 channels
  const float* A = adjs + (size_t)c * NN * NN;

  // cooperative A staging: thread covers rows (tid>>3)+32i, float4 slot (tid&7)
  const int r0 = tid >> 3;
  const int jj = tid & 7;

  const int half = lane >> 4;           // selects K sub-range per A/B layout
  const int rl   = lane & 15;

  v8f acc[8] = {};                      // 16 rows x 128 cols per wave

  const _Float16* xb_lane = xb + (size_t)lane * 16;

  // prologue: kick off DMA for tile 0
  stage_tile(A, ldsA, row0, r0, jj, 0);

  for (int kk = 0; kk < NITER; ++kk) {
    if (kk + 1 < NITER) {
      // issue next tile's DMA, then wait only for the *older* group:
      // each wave owns exactly 4 outstanding DMAs per tile; ASYNCcnt is in-order.
      stage_tile(A, ldsA, row0, r0, jj, kk + 1);
#ifdef USE_ASYNC_LDS
      __builtin_amdgcn_s_wait_asynccnt(4);
#endif
    } else {
#ifdef USE_ASYNC_LDS
      __builtin_amdgcn_s_wait_asynccnt(0);
#endif
    }
    __syncthreads();

    const int buf = (kk % 3) * (TM * APADF);

    // A fragment (ISA 16-bit A 16x32 layout): lane<16: K{0..7,16..23};
    // lane>=16: K{8..15,24..31}; convert f32->f16 at read.
    const float* ab = ldsA + buf + (w * 16 + rl) * APADF + half * 8;
    v4f a0 = *(const v4f*)ab;
    v4f a1 = *(const v4f*)(ab + 4);
    v4f a2 = *(const v4f*)(ab + 16);
    v4f a3 = *(const v4f*)(ab + 20);
    v16h afrag;
#pragma unroll
    for (int i = 0; i < 4; ++i) {
      afrag[i]      = (_Float16)a0[i];
      afrag[4 + i]  = (_Float16)a1[i];
      afrag[8 + i]  = (_Float16)a2[i];
      afrag[12 + i] = (_Float16)a3[i];
    }

    const _Float16* bb = xb_lane + (size_t)kk * (8 * 512);
#pragma unroll
    for (int nt = 0; nt < 8; ++nt) {
      v16h bfrag = *(const v16h*)(bb + nt * 512);
      acc[nt] = __builtin_amdgcn_wmma_f32_16x16x32_f16(
          false, afrag, false, bfrag, (short)0, acc[nt], false, false);
    }
  }
  __syncthreads();

  // write Y (f16) to LDS, converting C/D layout -> row-major for A-operand reads
#pragma unroll
  for (int nt = 0; nt < 8; ++nt) {
#pragma unroll
    for (int v = 0; v < 8; ++v) {
      int rloc = w * 16 + v + 8 * half;
      int colc = nt * 16 + rl;
      ldsY[rloc * YPAD + colc] = (_Float16)acc[nt][v];
    }
  }
  __syncthreads();

  // second GEMM: Z = Y @ adjW_c  (K = 128)
  v8f acc2[8] = {};
  const _Float16* wab_c = wab + (size_t)c * (4 * 8 * 512) + (size_t)lane * 16;
#pragma unroll
  for (int kk2 = 0; kk2 < 4; ++kk2) {
    const _Float16* yb = ldsY + (w * 16 + rl) * YPAD + kk2 * KT + half * 8;
    v8h lo = *(const v8h*)yb;
    v8h hi = *(const v8h*)(yb + 16);
    v16h a2v;
#pragma unroll
    for (int i = 0; i < 8; ++i) { a2v[i] = lo[i]; a2v[8 + i] = hi[i]; }
#pragma unroll
    for (int nt = 0; nt < 8; ++nt) {
      v16h bf = *(const v16h*)(wab_c + (kk2 * 8 + nt) * 512);
      acc2[nt] = __builtin_amdgcn_wmma_f32_16x16x32_f16(
          false, a2v, false, bf, (short)0, acc2[nt], false, false);
    }
  }

  // per-channel ReLU, accumulate across channels
#pragma unroll
  for (int nt = 0; nt < 8; ++nt) {
#pragma unroll
    for (int v = 0; v < 8; ++v) {
      float z = acc2[nt][v];
      z = z > 0.f ? z : 0.f;
      int rg   = row0 + w * 16 + v + 8 * half;
      int colc = nt * 16 + rl;
      atomicAdd(accbuf + (size_t)rg * FD + colc, z);
    }
  }
}

// ---------------------------------------------------------------------------
// Epilogue: out = relu(X @ W + b + accbuf)
// ---------------------------------------------------------------------------
__global__ __launch_bounds__(256) void xpart_epilogue(const float* __restrict__ X,
                                                      const _Float16* __restrict__ wb,
                                                      const float* __restrict__ bvec,
                                                      const float* __restrict__ accbuf,
                                                      float* __restrict__ out) {
  const int tid  = threadIdx.x;
  const int lane = tid & 31;
  const int w    = tid >> 5;
  const int half = lane >> 4;
  const int rl   = lane & 15;
  const int row0 = blockIdx.x * TM;
  const int gr   = row0 + w * 16 + rl;   // A-operand row for this lane

  v8f acc[8] = {};
  const _Float16* wlane = wb + (size_t)lane * 16;
#pragma unroll
  for (int kk2 = 0; kk2 < 4; ++kk2) {
    const float* xp = X + (size_t)gr * FD + kk2 * KT + half * 8;
    v4f f0 = *(const v4f*)xp;
    v4f f1 = *(const v4f*)(xp + 4);
    v4f f2 = *(const v4f*)(xp + 16);
    v4f f3 = *(const v4f*)(xp + 20);
    v16h a;
#pragma unroll
    for (int i = 0; i < 4; ++i) {
      a[i]      = (_Float16)f0[i];
      a[4 + i]  = (_Float16)f1[i];
      a[8 + i]  = (_Float16)f2[i];
      a[12 + i] = (_Float16)f3[i];
    }
#pragma unroll
    for (int nt = 0; nt < 8; ++nt) {
      v16h bf = *(const v16h*)(wlane + (kk2 * 8 + nt) * 512);
      acc[nt] = __builtin_amdgcn_wmma_f32_16x16x32_f16(
          false, a, false, bf, (short)0, acc[nt], false, false);
    }
  }

#pragma unroll
  for (int nt = 0; nt < 8; ++nt) {
#pragma unroll
    for (int v = 0; v < 8; ++v) {
      int rg   = row0 + w * 16 + v + 8 * half;
      int colc = nt * 16 + rl;
      size_t idx = (size_t)rg * FD + colc;
      float val = acc[nt][v] + accbuf[idx] + bvec[colc];
      out[idx] = val > 0.f ? val : 0.f;
    }
  }
}

// ---------------------------------------------------------------------------
extern "C" void kernel_launch(void* const* d_in, const int* in_sizes, int n_in,
                              void* d_out, int out_size, void* d_ws, size_t ws_size,
                              hipStream_t stream) {
  (void)in_sizes; (void)n_in; (void)out_size; (void)ws_size;

  const float* X    = (const float*)d_in[0];
  const float* adjs = (const float*)d_in[1];
  const float* W    = (const float*)d_in[2];
  const float* adjW = (const float*)d_in[3];
  const float* bvec = (const float*)d_in[4];
  float* out = (float*)d_out;

  // workspace layout (needs ~6.4 MB)
  char* ws = (char*)d_ws;
  _Float16* xb   = (_Float16*)ws;                               // 2 MB  (8192*128 f16, fragment-major)
  _Float16* wab  = (_Float16*)(ws + (size_t)(2u << 20));        // 128 KB (4x128x128 f16 fragments)
  _Float16* wbW  = (_Float16*)(ws + (size_t)(2u << 20) + (128u << 10)); // 32 KB
  float*    accb = (float*)  (ws + (size_t)(2u << 20) + (160u << 10));  // 4 MB f32 accumulator

  (void)hipMemsetAsync(accb, 0, (size_t)NN * FD * sizeof(float), stream);
  pack_x<<<256, 256, 0, stream>>>(X, xb);
  pack_w<<<16, 256, 0, stream>>>(adjW, wab);
  pack_w<<<4, 256, 0, stream>>>(W, wbW);

  dim3 g2(NN / TM, 4);   // 64 row tiles x 4 channels
  adj_gemm<<<g2, 256, 0, stream>>>(adjs, xb, wab, accb);
  xpart_epilogue<<<NN / TM, 256, 0, stream>>>(X, wbW, bvec, accb, out);
}
